// MambaG2G_6536940224932
// MI455X (gfx1250) — matrix operand order repaired and checked by reading, not compiled, not run on hardware
//
#include <hip/hip_runtime.h>
#include <hip/hip_bf16.h>
#include <cstdint>

// ---------------------------------------------------------------------------
// Types for CDNA5 WMMA (wave32)
// ---------------------------------------------------------------------------
typedef __bf16 bf16_t;
typedef __attribute__((ext_vector_type(8)))  __bf16 v8bf;
typedef __attribute__((ext_vector_type(16))) __bf16 v16bf;
typedef __attribute__((ext_vector_type(8)))  float  v8f;

// ---------------------------------------------------------------------------
// Problem constants (from reference)
// ---------------------------------------------------------------------------
#define DM    512          // d_model
#define DI    1024         // d_inner
#define NB    8            // batch
#define LLEN  1024         // seq len
#define BL    (NB * LLEN)  // 8192 rows
#define NST   16           // d_state
#define DTR   32           // dt_rank
#define PROJC 64           // dt_rank + 2*d_state
#define DIN   1899
#define DOUT  256
#define SCHUNK 16          // scan timesteps per async-staged LDS chunk

// ---------------------------------------------------------------------------
// Helpers
// ---------------------------------------------------------------------------
__device__ __forceinline__ bf16_t f2bf(float f) {
  union { float f; unsigned u; } v; v.f = f;
  unsigned r = v.u + 0x7FFFu + ((v.u >> 16) & 1u);   // round-to-nearest-even
  union { unsigned short s; bf16_t b; } o; o.s = (unsigned short)(r >> 16);
  return o.b;
}
__device__ __forceinline__ float sigmoidf_(float x) { return 1.0f / (1.0f + __expf(-x)); }
__device__ __forceinline__ float softplusf_(float x) { return (x > 20.0f) ? x : log1pf(__expf(x)); }
__device__ __forceinline__ float eluf_(float x) { return (x > 0.0f) ? x : expm1f(x); }

__device__ __forceinline__ v8f wmma_bf16(v16bf a, v16bf b, v8f c) {
  // (neg_a, A, neg_b, B, c_mod, C, reuse_a, reuse_b)
  return __builtin_amdgcn_wmma_f32_16x16x32_bf16(false, a, false, b, (short)0, c,
                                                 false, false);
}

// A-fragment: 16x32 bf16, M x K row-major source.
// lane m=lane&15; lanes 0-15 hold K[k0..k0+15];
// lanes 16-31 hold K[k0+16..k0+31].
__device__ __forceinline__ v16bf load_a_frag(const bf16_t* __restrict__ A, int lda,
                                             int m0, int k0, int lane, int M) {
  int row = m0 + (lane & 15);
  row = row < M ? row : M - 1;
  const bf16_t* p = A + (size_t)row * lda + k0 + ((lane >> 4) << 4);
  v8bf lo = *(const v8bf*)(p);
  v8bf hi = *(const v8bf*)(p + 8);
  return __builtin_shufflevector(lo, hi, 0, 1, 2, 3, 4, 5, 6, 7,
                                 8, 9, 10, 11, 12, 13, 14, 15);
}

// B-fragment: 32x16 bf16 from weights stored N x K row-major (W[n][k]).
// lane n=lane&15 holds column n; lanes 0-15 -> K[k0..k0+15], 16-31 -> K[k0+16..k0+31].
__device__ __forceinline__ v16bf load_b_frag(const bf16_t* __restrict__ W, int ldw,
                                             int n0, int k0, int lane, int N) {
  int col = n0 + (lane & 15);
  col = col < N ? col : N - 1;
  const bf16_t* p = W + (size_t)col * ldw + k0 + ((lane >> 4) << 4);
  return *(const v16bf*)(p);
}

// ---------------------------------------------------------------------------
// f32 -> bf16 conversion / zero
// ---------------------------------------------------------------------------
__global__ __launch_bounds__(256) void f32_to_bf16_kernel(const float* __restrict__ s,
                                                          bf16_t* __restrict__ d, int n) {
  int i = blockIdx.x * 256 + threadIdx.x;
  if (i < n) d[i] = f2bf(s[i]);
}

__global__ __launch_bounds__(256) void zero_f32_kernel(float* __restrict__ p, int n) {
  int i = blockIdx.x * 256 + threadIdx.x;
  if (i < n) p[i] = 0.0f;
}

// ---------------------------------------------------------------------------
// WMMA GEMM: C[M,N] = A[M,K](bf16, row-major) * W[N,K](bf16, row-major)^T
// 256 threads = 8 waves; wave tile 32x32 (2x2 accumulators).
// mode 0: store f32 C
// mode 1: store f32 C and bf16 copy of cols < DTR into bfout (ld = DTR)
// mode 2: C = softplus(acc + bias[col])
// mode 3: atomicAdd(accum[(row>>rshift)*N + col], acc * scale)   (mean over L)
// ---------------------------------------------------------------------------
__global__ __launch_bounds__(256) void wmma_gemm_bf16_kernel(
    const bf16_t* __restrict__ A, int lda,
    const bf16_t* __restrict__ W, int ldw,
    float* __restrict__ C, int ldc,
    int M, int N, int K,
    int wavesM, int mode,
    const float* __restrict__ bias,
    bf16_t* __restrict__ bfout,
    float* __restrict__ accum,
    int rshift, float scale) {
  int wave = threadIdx.x >> 5;
  int lane = threadIdx.x & 31;
  int wm = wave % wavesM;
  int wn = wave / wavesM;
  int m0 = blockIdx.x * (wavesM * 32) + wm * 32;
  int n0 = blockIdx.y * ((8 / wavesM) * 32) + wn * 32;

  v8f acc[2][2] = {};
  for (int k0 = 0; k0 < K; k0 += 32) {
    if (k0 + 64 < K) {  // speculative L2 prefetch of next-next A tile (global_prefetch_b8)
      __builtin_prefetch(A + (size_t)(m0 + (lane & 15)) * lda + k0 + 64, 0, 1);
    }
    v16bf a0 = load_a_frag(A, lda, m0, k0, lane, M);
    v16bf a1 = load_a_frag(A, lda, m0 + 16, k0, lane, M);
    v16bf b0 = load_b_frag(W, ldw, n0, k0, lane, N);
    v16bf b1 = load_b_frag(W, ldw, n0 + 16, k0, lane, N);
    acc[0][0] = wmma_bf16(a0, b0, acc[0][0]);
    acc[0][1] = wmma_bf16(a0, b1, acc[0][1]);
    acc[1][0] = wmma_bf16(a1, b0, acc[1][0]);
    acc[1][1] = wmma_bf16(a1, b1, acc[1][1]);
  }

#pragma unroll
  for (int ti = 0; ti < 2; ++ti) {
#pragma unroll
    for (int tj = 0; tj < 2; ++tj) {
      int col = n0 + tj * 16 + (lane & 15);
      int rbase = m0 + ti * 16 + ((lane >> 4) << 3);
#pragma unroll
      for (int r = 0; r < 8; ++r) {
        int row = rbase + r;
        if (row >= M || col >= N) continue;
        float val = acc[ti][tj][r];
        if (mode == 0) {
          C[(size_t)row * ldc + col] = val;
        } else if (mode == 1) {
          C[(size_t)row * ldc + col] = val;
          if (col < DTR) bfout[(size_t)row * DTR + col] = f2bf(val);
        } else if (mode == 2) {
          C[(size_t)row * ldc + col] = softplusf_(val + bias[col]);
        } else {
          atomicAdd(&accum[(size_t)(row >> rshift) * N + col], val * scale);
        }
      }
    }
  }
}

// ---------------------------------------------------------------------------
// Causal depthwise conv (D_CONV=4) + bias + SiLU, fused. Emits bf16 (x_proj A
// operand) and f32 (scan input u).
// ---------------------------------------------------------------------------
__global__ __launch_bounds__(256) void conv_silu_kernel(
    const float* __restrict__ xz, const float* __restrict__ cw,
    const float* __restrict__ cb, bf16_t* __restrict__ xi_bf,
    float* __restrict__ u) {
  int idx = blockIdx.x * 256 + threadIdx.x;  // over BL*DI
  if (idx >= BL * DI) return;
  int c = idx & (DI - 1);
  int bl = idx >> 10;
  int l = bl & (LLEN - 1);
  float acc = cb[c];
#pragma unroll
  for (int j = 0; j < 4; ++j) {
    int ls = l - 3 + j;
    if (ls >= 0) acc += cw[c * 4 + j] * xz[(size_t)(bl - 3 + j) * (2 * DI) + c];
  }
  float s = acc * sigmoidf_(acc);
  xi_bf[(size_t)bl * DI + c] = f2bf(s);
  u[(size_t)bl * DI + c] = s;
}

// ---------------------------------------------------------------------------
// Selective scan with async-to-LDS double-buffered staging of proj rows.
// One thread per (b, d) channel, h[16] in VGPRs, sequential over L.
// The wave-uniform B_t/C_t vectors are DMA'd to LDS one SCHUNK ahead with
// GLOBAL_LOAD_ASYNC_TO_LDS_B128 (ASYNCcnt) so the serial dependence chain
// reads them via low-latency ds_load instead of an L2 round trip.
// Fused with D-skip and silu(z) gating; emits bf16 out_proj A operand.
// ---------------------------------------------------------------------------
__global__ __launch_bounds__(256) void scan_kernel(
    const float* __restrict__ dt, const float* __restrict__ u,
    const float* __restrict__ xz, const float* __restrict__ proj,
    const float* __restrict__ Alog, const float* __restrict__ Dp,
    bf16_t* __restrict__ yg) {
  __shared__ float pbuf[2][SCHUNK * PROJC];  // 2 x 4KB double buffer

  int tid = threadIdx.x;
  int d = blockIdx.x * 256 + tid;
  int b = blockIdx.y;

  // Async chunk preload: 4KB chunk = 256 threads x 16B, one b128 DMA per lane.
  const char* gbase = (const char*)(proj + (size_t)b * LLEN * PROJC);
  unsigned lds_base0 = (unsigned)(uintptr_t)(&pbuf[0][0]);  // flat addr[31:0] == LDS offset
  unsigned lds_base1 = (unsigned)(uintptr_t)(&pbuf[1][0]);
  auto preload = [&](int chunk, int bufi) {
    unsigned lds = (bufi ? lds_base1 : lds_base0) + (unsigned)tid * 16u;
    unsigned long long g =
        (unsigned long long)(gbase + (size_t)chunk * (SCHUNK * PROJC * 4) + (size_t)tid * 16);
    asm volatile("global_load_async_to_lds_b128 %0, %1, off"
                 :: "v"(lds), "v"(g)
                 : "memory");
  };

  preload(0, 0);

  float Av[NST], h[NST];
#pragma unroll
  for (int n = 0; n < NST; ++n) {
    Av[n] = -__expf(Alog[d * NST + n]);
    h[n] = 0.0f;
  }
  float Dv = Dp[d];

  const int nchunks = LLEN / SCHUNK;
  for (int c = 0; c < nchunks; ++c) {
    int bufi = c & 1;
    asm volatile("s_wait_asynccnt 0x0" ::: "memory");  // chunk c landed in LDS
    __syncthreads();                                    // publish to all waves
    if (c + 1 < nchunks) preload(c + 1, bufi ^ 1);      // overlap DMA with compute

    const float* pb = &pbuf[bufi][0];
#pragma unroll 4
    for (int li = 0; li < SCHUNK; ++li) {
      int l = c * SCHUNK + li;
      size_t row = (size_t)b * LLEN + l;
      if (l + SCHUNK < LLEN) {  // warm L2 for next chunk's per-lane streams
        __builtin_prefetch(&dt[(row + SCHUNK) * DI + d], 0, 1);
        __builtin_prefetch(&u[(row + SCHUNK) * DI + d], 0, 1);
        __builtin_prefetch(&xz[(row + SCHUNK) * (2 * DI) + DI + d], 0, 1);
      }
      float dtv = dt[row * DI + d];
      float uv = u[row * DI + d];
      const float* pc = pb + li * PROJC;  // wave-uniform LDS reads (ds_load broadcast)
      float dtu = dtv * uv;
      float y = 0.0f;
#pragma unroll
      for (int n = 0; n < NST; ++n) {
        float dA = __expf(dtv * Av[n]);
        h[n] = dA * h[n] + dtu * pc[DTR + n];   // B_t
        y = fmaf(h[n], pc[DTR + NST + n], y);   // C_t
      }
      float zv = xz[row * (2 * DI) + DI + d];
      float yo = (y + uv * Dv) * (zv * sigmoidf_(zv));
      yg[row * DI + d] = f2bf(yo);
    }
    __syncthreads();  // all waves done reading bufi before it is re-filled
  }
}

// ---------------------------------------------------------------------------
// Head: x = elu(tanh(e @ out_fc_w^T + b)), then mu/sigma FCs.
// ---------------------------------------------------------------------------
__global__ __launch_bounds__(128) void head_fc1_kernel(
    const float* __restrict__ e, const float* __restrict__ w,
    const float* __restrict__ bias, float* __restrict__ xh,
    float* __restrict__ out) {
  int i = blockIdx.x * 128 + threadIdx.x;
  int b = blockIdx.y;
  if (i >= DIN) return;
  const float* ev = e + (size_t)b * DM;
  const float* wr = w + (size_t)i * DM;
  float acc = bias[i];
  for (int k = 0; k < DM; ++k) acc = fmaf(ev[k], wr[k], acc);
  float xo = eluf_(tanhf(acc));
  xh[(size_t)b * DIN + i] = xo;
  out[(size_t)b * DIN + i] = xo;
}

__global__ __launch_bounds__(128) void head_fc2_kernel(
    const float* __restrict__ xh, const float* __restrict__ mw,
    const float* __restrict__ mb, const float* __restrict__ sw,
    const float* __restrict__ sb, float* __restrict__ out) {
  int o = blockIdx.x * 128 + threadIdx.x;
  int b = blockIdx.y;
  if (o >= DOUT) return;
  const float* xr = xh + (size_t)b * DIN;
  const float* mr = mw + (size_t)o * DIN;
  const float* sr = sw + (size_t)o * DIN;
  float am = mb[o], as = sb[o];
  for (int k = 0; k < DIN; ++k) {
    am = fmaf(xr[k], mr[k], am);
    as = fmaf(xr[k], sr[k], as);
  }
  float* mu = out + (size_t)NB * DIN;
  float* sg = mu + (size_t)NB * DOUT;
  mu[(size_t)b * DOUT + o] = am;
  sg[(size_t)b * DOUT + o] = eluf_(as) + 1.0f + 1e-14f;
}

// ---------------------------------------------------------------------------
// Launch
// ---------------------------------------------------------------------------
extern "C" void kernel_launch(void* const* d_in, const int* in_sizes, int n_in,
                              void* d_out, int out_size, void* d_ws, size_t ws_size,
                              hipStream_t stream) {
  const float* input      = (const float*)d_in[0];
  const float* in_proj_w  = (const float*)d_in[1];
  const float* conv_w     = (const float*)d_in[2];
  const float* conv_b     = (const float*)d_in[3];
  const float* x_proj_w   = (const float*)d_in[4];
  const float* dt_proj_w  = (const float*)d_in[5];
  const float* dt_proj_b  = (const float*)d_in[6];
  const float* A_log      = (const float*)d_in[7];
  const float* D_param    = (const float*)d_in[8];
  const float* out_proj_w = (const float*)d_in[9];
  const float* out_fc_w   = (const float*)d_in[10];
  const float* out_fc_b   = (const float*)d_in[11];
  const float* mu_fc_w    = (const float*)d_in[12];
  const float* mu_fc_b    = (const float*)d_in[13];
  const float* sigma_fc_w = (const float*)d_in[14];
  const float* sigma_fc_b = (const float*)d_in[15];
  float* out = (float*)d_out;

  char* ws = (char*)d_ws;
  size_t off = 0;
  auto alloc = [&](size_t bytes) -> void* {
    void* p = ws + off;
    off = (off + bytes + 255) & ~(size_t)255;
    return p;
  };

  bf16_t* Xb    = (bf16_t*)alloc((size_t)BL * DM * 2);
  bf16_t* Wb1   = (bf16_t*)alloc((size_t)2 * DI * DM * 2);
  bf16_t* Wbx   = (bf16_t*)alloc((size_t)PROJC * DI * 2);
  bf16_t* Wbdt  = (bf16_t*)alloc((size_t)DI * DTR * 2);
  bf16_t* Wbo   = (bf16_t*)alloc((size_t)DM * DI * 2);
  float*  XZ    = (float*)alloc((size_t)BL * 2 * DI * 4);
  bf16_t* xi_bf = (bf16_t*)alloc((size_t)BL * DI * 2);
  float*  ubuf  = (float*)alloc((size_t)BL * DI * 4);
  float*  proj  = (float*)alloc((size_t)BL * PROJC * 4);
  bf16_t* dtraw = (bf16_t*)alloc((size_t)BL * DTR * 2);
  float*  dtbuf = (float*)alloc((size_t)BL * DI * 4);
  bf16_t* yg    = (bf16_t*)alloc((size_t)BL * DI * 2);
  float*  ebuf  = (float*)alloc((size_t)NB * DM * 4);
  float*  xh    = (float*)alloc((size_t)NB * DIN * 4);
  (void)ws_size; (void)in_sizes; (void)n_in; (void)out_size;

  auto cvt = [&](const float* s, bf16_t* d, int n) {
    f32_to_bf16_kernel<<<(n + 255) / 256, 256, 0, stream>>>(s, d, n);
  };
  cvt(input,      Xb,   BL * DM);
  cvt(in_proj_w,  Wb1,  2 * DI * DM);
  cvt(x_proj_w,   Wbx,  PROJC * DI);
  cvt(dt_proj_w,  Wbdt, DI * DTR);
  cvt(out_proj_w, Wbo,  DM * DI);

  // in_proj: XZ[8192,2048] = X * W^T    (waves 2x4 -> block tile 64x128)
  wmma_gemm_bf16_kernel<<<dim3(BL / 64, (2 * DI) / 128), 256, 0, stream>>>(
      Xb, DM, Wb1, DM, XZ, 2 * DI, BL, 2 * DI, DM,
      2, 0, nullptr, nullptr, nullptr, 0, 0.0f);

  // depthwise causal conv + SiLU
  conv_silu_kernel<<<(BL * DI) / 256, 256, 0, stream>>>(XZ, conv_w, conv_b, xi_bf, ubuf);

  // x_proj: proj[8192,64]   (waves 8x1 -> block tile 256x32); also emits bf16 dt_raw
  wmma_gemm_bf16_kernel<<<dim3(BL / 256, PROJC / 32), 256, 0, stream>>>(
      xi_bf, DI, Wbx, DI, proj, PROJC, BL, PROJC, DI,
      8, 1, nullptr, dtraw, nullptr, 0, 0.0f);

  // dt_proj + bias + softplus: dt[8192,1024]
  wmma_gemm_bf16_kernel<<<dim3(BL / 64, DI / 128), 256, 0, stream>>>(
      dtraw, DTR, Wbdt, DTR, dtbuf, DI, BL, DI, DTR,
      2, 2, dt_proj_b, nullptr, nullptr, 0, 0.0f);

  // selective scan (async-to-LDS staged B_t/C_t) + D skip + silu(z) gate -> yg
  scan_kernel<<<dim3(DI / 256, NB), 256, 0, stream>>>(
      dtbuf, ubuf, XZ, proj, A_log, D_param, yg);

  // out_proj with fused mean over L (atomic accumulate into e)
  zero_f32_kernel<<<(NB * DM + 255) / 256, 256, 0, stream>>>(ebuf, NB * DM);
  wmma_gemm_bf16_kernel<<<dim3(BL / 64, DM / 128), 256, 0, stream>>>(
      yg, DI, Wbo, DI, nullptr, 0, BL, DM, DI,
      2, 3, nullptr, nullptr, ebuf, 10 /* >>10 == /L */, 1.0f / (float)LLEN);

  // head
  head_fc1_kernel<<<dim3((DIN + 127) / 128, NB), 128, 0, stream>>>(
      ebuf, out_fc_w, out_fc_b, xh, out);
  head_fc2_kernel<<<dim3((DOUT + 127) / 128, NB), 128, 0, stream>>>(
      xh, mu_fc_w, mu_fc_b, sigma_fc_w, sigma_fc_b, out);
}
